// T_ALIGNNAtomWise_25082609008768
// MI455X (gfx1250) — matrix operand order repaired and checked by reading, not compiled, not run on hardware
//
#include <hip/hip_runtime.h>

typedef __bf16 bf16;
typedef __attribute__((ext_vector_type(16))) __bf16 v16bf;
typedef __attribute__((ext_vector_type(8)))  float  v8f;
typedef __attribute__((ext_vector_type(4)))  float  v4f;

#define MODE_BIAS        0
#define MODE_LN_SILU     1
#define MODE_ADD_LNSILU  2

__device__ __forceinline__ float silu_f(float x) { return x / (1.f + __expf(-x)); }

// ---------------------------------------------------------------------------
// Pack a [K,64] f32 weight into bf16 WMMA B-fragment order, K padded to Kpad.
// dst[((t*64 + n)*32) + kk] = W[(t*32+kk), n]  (zero pad kk beyond K), so a
// lane's 16 bf16 (8 VGPRs, K=2v(+16*hi)) are one contiguous 32B chunk.
// ---------------------------------------------------------------------------
__global__ void pack_w_kernel(const float* __restrict__ W, int K,
                              bf16* __restrict__ dst, int total) {
  int i = blockIdx.x * blockDim.x + threadIdx.x;
  if (i >= total) return;
  int kk   = i & 31;
  int rest = i >> 5;
  int n    = rest & 63;
  int t    = rest >> 6;
  int kg   = t * 32 + kk;
  float v  = (kg < K) ? W[kg * 64 + n] : 0.f;
  dst[i] = (bf16)v;
}

__global__ void zero_kernel(float* __restrict__ p, int n) {
  int i = blockIdx.x * blockDim.x + threadIdx.x;
  if (i < n) p[i] = 0.f;
}

// Pad-copy atom features [N,92] -> [N,96] (zeros in cols 92..95)
__global__ void pad_atom_kernel(const float* __restrict__ in,
                                float* __restrict__ out, int N) {
  int i = blockIdx.x * blockDim.x + threadIdx.x;
  int total = N * 96;
  if (i >= total) return;
  int row = i / 96, col = i % 96;
  out[i] = (col < 92) ? in[row * 92 + col] : 0.f;
}

// ---------------------------------------------------------------------------
// RBF featurization kernels — write rows zero-padded to the GEMM K stride.
// ---------------------------------------------------------------------------
__global__ void rbf_edge_kernel(const float* __restrict__ r,
                                float* __restrict__ out, int E) {
  int i = blockIdx.x * blockDim.x + threadIdx.x;
  int total = E * 96;
  if (i >= total) return;
  int e = i / 96, b = i % 96;
  if (b >= 80) { out[i] = 0.f; return; }
  float x = r[e * 3 + 0], y = r[e * 3 + 1], z = r[e * 3 + 2];
  float d = sqrtf(x * x + y * y + z * z);
  const float step = 8.f / 79.f;
  const float gamma = 1.f / (step * step);
  float t = d - step * (float)b;
  out[i] = __expf(-gamma * t * t);
}

__global__ void rbf_angle_kernel(const float* __restrict__ r,
                                 const int* __restrict__ ls,
                                 const int* __restrict__ ld,
                                 float* __restrict__ out, int L) {
  int i = blockIdx.x * blockDim.x + threadIdx.x;
  int total = L * 64;
  if (i >= total) return;
  int l = i / 64, b = i % 64;
  if (b >= 40) { out[i] = 0.f; return; }
  int a = ls[l], c = ld[l];
  float ax = -r[a * 3 + 0], ay = -r[a * 3 + 1], az = -r[a * 3 + 2];
  float bx =  r[c * 3 + 0], by =  r[c * 3 + 1], bz =  r[c * 3 + 2];
  float dot = ax * bx + ay * by + az * bz;
  float na = sqrtf(ax * ax + ay * ay + az * az);
  float nb = sqrtf(bx * bx + by * by + bz * bz);
  float cosv = dot / (na * nb);
  cosv = fminf(1.f, fmaxf(-1.f, cosv));
  const float step = 2.f / 39.f;
  const float gamma = 1.f / (step * step);
  float t = cosv - (-1.f + step * (float)b);
  out[i] = __expf(-gamma * t * t);
}

// ---------------------------------------------------------------------------
// Fused GEMM: out[M,64] = epilogue( X[M,K] @ W[K,64] + bias )
// Requirements (guaranteed by caller): M % 16 == 0, K % 32 == 0, X stride = K.
// Block = 256 threads = 8 waves; each wave owns 16 rows x 64 cols
// (4 accumulators, 4 WMMAs per K-tile, A fragment loaded once per tile).
// Guard is wave-uniform only. Epilogue LN stats via width-16 shfl_xor:
//   MODE_BIAS       : out = g
//   MODE_LN_SILU    : out = silu(LN(g))
//   MODE_ADD_LNSILU : out = g + silu(LN(extra))     (EGC residual update)
// ---------------------------------------------------------------------------
__global__ __launch_bounds__(256) void gemm64_wmma_kernel(
    const float* __restrict__ X, int M, int K,
    const bf16* __restrict__ Wp, const float* __restrict__ bias,
    const float* __restrict__ lng, const float* __restrict__ lnb,
    const float* __restrict__ extra, float* __restrict__ out, int mode) {
  const int tid  = threadIdx.x;
  const int wave = tid >> 5;
  const int lane = tid & 31;
  const int hi   = lane >> 4;
  const int l15  = lane & 15;
  const int rowT = blockIdx.x * 128 + wave * 16;  // wave-uniform tile base
  if (rowT >= M) return;                          // uniform branch, no divergence

  v8f acc[4];
#pragma unroll
  for (int n = 0; n < 4; ++n)
    acc[n] = (v8f){0.f, 0.f, 0.f, 0.f, 0.f, 0.f, 0.f, 0.f};

  const float* xr = X + (size_t)(rowT + l15) * (size_t)K;
  const int T = K >> 5;
  for (int t = 0; t < T; ++t) {
    // A fragment 16x32 bf16: elems 0..15 <- K=16*hi+0..15
    const int b0 = t * 32 + hi * 16;
    v4f f0 = *(const v4f*)(xr + b0);
    v4f f1 = *(const v4f*)(xr + b0 + 4);
    v4f f2 = *(const v4f*)(xr + b0 + 8);
    v4f f3 = *(const v4f*)(xr + b0 + 12);
    v16bf a;
#pragma unroll
    for (int j = 0; j < 4; ++j) {
      a[j]      = (bf16)f0[j];
      a[4 + j]  = (bf16)f1[j];
      a[8 + j]  = (bf16)f2[j];
      a[12 + j] = (bf16)f3[j];
    }
#pragma unroll
    for (int n = 0; n < 4; ++n) {
      const v16bf b =
          *(const v16bf*)(Wp + (((t * 64 + n * 16 + l15) << 5) + hi * 16));
      acc[n] = __builtin_amdgcn_wmma_f32_16x16x32_bf16(false, a, false, b,
                                                       (short)0, acc[n], false, false);
    }
  }

  // ---- Epilogue: lane holds rows rowT+8*hi+r (r=0..7), cols n*16+l15 -------
  float bia[4], gg[4], bb[4];
#pragma unroll
  for (int n = 0; n < 4; ++n) bia[n] = bias[n * 16 + l15];
  if (mode != MODE_BIAS) {
#pragma unroll
    for (int n = 0; n < 4; ++n) {
      gg[n] = lng[n * 16 + l15];
      bb[n] = lnb[n * 16 + l15];
    }
  }

  float vals[4][8];
#pragma unroll
  for (int n = 0; n < 4; ++n)
#pragma unroll
    for (int rr = 0; rr < 8; ++rr) vals[n][rr] = acc[n][rr] + bia[n];

  if (mode == MODE_BIAS) {
#pragma unroll
    for (int rr = 0; rr < 8; ++rr) {
      const size_t ro = (size_t)(rowT + hi * 8 + rr) * 64 + l15;
#pragma unroll
      for (int n = 0; n < 4; ++n) out[ro + n * 16] = vals[n][rr];
    }
    return;
  }

  float ext[4][8];
  if (mode == MODE_ADD_LNSILU) {
#pragma unroll
    for (int rr = 0; rr < 8; ++rr) {
      const size_t ro = (size_t)(rowT + hi * 8 + rr) * 64 + l15;
#pragma unroll
      for (int n = 0; n < 4; ++n) ext[n][rr] = extra[ro + n * 16];
    }
  }

  // per-row LN stats: reduce across the 16 lanes of this half-wave
  float mu[8], rs[8];
#pragma unroll
  for (int rr = 0; rr < 8; ++rr) {
    float s = 0.f, s2 = 0.f;
#pragma unroll
    for (int n = 0; n < 4; ++n) {
      float bse = (mode == MODE_ADD_LNSILU) ? ext[n][rr] : vals[n][rr];
      s += bse;
      s2 += bse * bse;
    }
#pragma unroll
    for (int off = 1; off < 16; off <<= 1) {
      s  += __shfl_xor(s, off, 16);
      s2 += __shfl_xor(s2, off, 16);
    }
    float m_ = s * (1.f / 64.f);
    float v_ = s2 * (1.f / 64.f) - m_ * m_;
    mu[rr] = m_;
    rs[rr] = rsqrtf(v_ + 1e-5f);
  }

#pragma unroll
  for (int rr = 0; rr < 8; ++rr) {
    const size_t ro = (size_t)(rowT + hi * 8 + rr) * 64 + l15;
#pragma unroll
    for (int n = 0; n < 4; ++n) {
      float sv = (mode == MODE_LN_SILU) ? vals[n][rr] : ext[n][rr];
      float nv = (sv - mu[rr]) * rs[rr] * gg[n] + bb[n];
      float ac = silu_f(nv);
      out[ro + n * 16] = (mode == MODE_LN_SILU) ? ac : (vals[n][rr] + ac);
    }
  }
}

// ---------------------------------------------------------------------------
// Edge-gated multi-head attention + scatter:
//   score[h] = <q[src],k[dst]>_head / sqrt(16);  m = score + e (in place)
//   attn = softmax over 4 heads per channel;  agg[dst] += v[src]*attn
// 256 threads = 4 edges x 64 channels; head dot via 16-lane shfl_xor (wave32).
// ---------------------------------------------------------------------------
__global__ __launch_bounds__(256) void egc_attn_kernel(
    const float* __restrict__ q, const float* __restrict__ k,
    const float* __restrict__ v, float* __restrict__ m,
    const int* __restrict__ src, const int* __restrict__ dst,
    int E, float* __restrict__ agg) {
  __shared__ float sm[4][64];
  const int tid = threadIdx.x;
  const int eg  = tid >> 6;
  const int c   = tid & 63;
  const int e   = blockIdx.x * 4 + eg;
  const bool ok = e < E;
  const int s = ok ? src[e] : 0;
  const int d = ok ? dst[e] : 0;

  float qv = ok ? q[(size_t)s * 64 + c] : 0.f;
  float kv = ok ? k[(size_t)d * 64 + c] : 0.f;
  float p = qv * kv;
#pragma unroll
  for (int off = 1; off < 16; off <<= 1) p += __shfl_xor(p, off, 16);
  float score = p * 0.25f;  // 1/sqrt(C), C=16
  float mv = score + (ok ? m[(size_t)e * 64 + c] : 0.f);
  sm[eg][c] = mv;
  __syncthreads();
  const int ci = c & 15;
  float m0 = sm[eg][ci], m1 = sm[eg][16 + ci], m2 = sm[eg][32 + ci], m3 = sm[eg][48 + ci];
  float mx = fmaxf(fmaxf(m0, m1), fmaxf(m2, m3));
  float den = __expf(m0 - mx) + __expf(m1 - mx) + __expf(m2 - mx) + __expf(m3 - mx);
  float attn = __expf(mv - mx) / den;
  if (ok) {
    m[(size_t)e * 64 + c] = mv;
    float msg = v[(size_t)s * 64 + c] * attn;
    atomicAdd(&agg[(size_t)d * 64 + c], msg);
  }
}

// ---------------------------------------------------------------------------
// Mean-pool partial sums (atomic into h[64]) and final scalar head
// ---------------------------------------------------------------------------
__global__ void pool_kernel(const float* __restrict__ x, float* __restrict__ h, int N) {
  const int c = threadIdx.x & 63;
  int row = blockIdx.x * (blockDim.x >> 6) + (threadIdx.x >> 6);
  const int rs = gridDim.x * (blockDim.x >> 6);
  float s = 0.f;
  for (; row < N; row += rs) s += x[(size_t)row * 64 + c];
  atomicAdd(&h[c], s);
}

__global__ void final_kernel(const float* __restrict__ h, const float* __restrict__ fw,
                             const float* __restrict__ fb, float* __restrict__ out,
                             float invN) {
  __shared__ float sr[64];
  const int c = threadIdx.x;
  sr[c] = h[c] * invN * fw[c];
  __syncthreads();
  for (int off = 32; off > 0; off >>= 1) {
    if (c < off) sr[c] += sr[c + off];
    __syncthreads();
  }
  if (c == 0) out[0] = sr[0] + fb[0];
}

// ---------------------------------------------------------------------------
// Host orchestration
// ---------------------------------------------------------------------------
extern "C" void kernel_launch(void* const* d_in, const int* in_sizes, int n_in,
                              void* d_out, int out_size, void* d_ws, size_t ws_size,
                              hipStream_t stream) {
  (void)in_sizes; (void)n_in; (void)out_size; (void)ws_size;
  const int N = 20000, E = 240000, L = 640000;

  auto F = [&](int i) -> const float* { return (const float*)d_in[i]; };
  const float* atomf = F(0);
  const float* r     = F(1);
  const int* src  = (const int*)d_in[120];
  const int* dst  = (const int*)d_in[121];
  const int* lsrc = (const int*)d_in[122];
  const int* ldst = (const int*)d_in[123];

  float* ws = (float*)d_ws;
  const size_t NF = (size_t)N * 64, EF = (size_t)E * 64, LF = (size_t)L * 64;
  float* X   = ws;
  float* AQ  = X + NF;
  float* AK  = AQ + NF;
  float* AV  = AK + NF;
  float* AGG = AV + NF;
  float* B1  = AGG + NF;      // y / m_out / y_new
  float* B2  = B1 + EF;       // e,m (node EGC) / q (line EGC)
  float* B4  = B2 + EF;       // k (line EGC)
  float* B5  = B4 + EF;       // v (line EGC)
  float* B6  = B5 + EF;       // agg (line EGC)
  float* C1  = B6 + EF;       // z (in-place updated)
  float* C2  = C1 + LF;       // rbf scratch / e,m (line EGC)
  float* H   = C2 + LF;       // 64 pooled floats
  float* XP  = H + 64;        // padded atom features [N,96]
  bf16*  PKB = (bf16*)(XP + (size_t)N * 96);

  // bf16 weight pack slots
  const int PK_ATOM = 0, PK_EEMB1 = 6144, PK_EEMB2 = 12288,
            PK_AEMB1 = 16384, PK_AEMB2 = 20480, PK_EGC = 24576;
  auto packw = [&](const float* W, int K, int Kpad, int off) {
    int total = Kpad * 64;
    pack_w_kernel<<<(total + 255) / 256, 256, 0, stream>>>(W, K, PKB + off, total);
  };
  // param leaf indices (jax pytree order: sorted dict keys; lists in order)
  packw(F(74), 92, 96, PK_ATOM);    // atom_emb.lin.W
  packw(F(78), 80, 96, PK_EEMB1);   // edge_emb1.lin.W
  packw(F(82), 64, 64, PK_EEMB2);   // edge_emb2.lin.W
  packw(F(66), 40, 64, PK_AEMB1);   // angle_emb1.lin.W
  packw(F(70), 64, 64, PK_AEMB2);   // angle_emb2.lin.W
  // EGC bases: a0.node, a0.edge, a1.node, a1.edge, gcn0, gcn1
  const int egc_base[6] = {18, 2, 50, 34, 88, 104};
  const int wslot[6] = {12, 4, 14, 0, 2, 10};  // q,k,v,e,ev,out W offsets
  for (int i = 0; i < 6; ++i)
    for (int j = 0; j < 6; ++j)
      packw(F(egc_base[i] + wslot[j]), 64, 64, PK_EGC + (i * 6 + j) * 4096);

  auto zero = [&](float* p, size_t n) {
    zero_kernel<<<(unsigned)((n + 255) / 256), 256, 0, stream>>>(p, (int)n);
  };
  zero(H, 64);

  auto gemm = [&](const float* Xp, int M, int K, const bf16* Wp,
                  const float* bias, const float* g, const float* b,
                  const float* extra, float* out, int mode) {
    gemm64_wmma_kernel<<<(M + 127) / 128, 256, 0, stream>>>(
        Xp, M, K, Wp, bias, g, b, extra, out, mode);
  };

  // ---- Embeddings -------------------------------------------------------
  rbf_edge_kernel<<<(E * 96 + 255) / 256, 256, 0, stream>>>(r, C2, E);
  gemm(C2, E, 96, PKB + PK_EEMB1, F(79), F(81), F(80), nullptr, B1, MODE_LN_SILU);
  gemm(B1, E, 64, PKB + PK_EEMB2, F(83), F(85), F(84), nullptr, B1, MODE_LN_SILU);
  rbf_angle_kernel<<<(L * 64 + 255) / 256, 256, 0, stream>>>(r, lsrc, ldst, C2, L);
  gemm(C2, L, 64, PKB + PK_AEMB1, F(67), F(69), F(68), nullptr, C1, MODE_LN_SILU);
  gemm(C1, L, 64, PKB + PK_AEMB2, F(71), F(73), F(72), nullptr, C1, MODE_LN_SILU);
  pad_atom_kernel<<<(N * 96 + 255) / 256, 256, 0, stream>>>(atomf, XP, N);
  gemm(XP, N, 96, PKB + PK_ATOM, F(75), F(77), F(76), nullptr, X, MODE_LN_SILU);

  // ---- EdgeGatedGraphConv ----------------------------------------------
  auto run_egc = [&](int pb, int ei, float* nf, float* ef,
                     const int* gs, const int* gd, int Nn, int Ne,
                     float* Q, float* Kf, float* V, float* Mb, float* Agg) {
    const bf16* pq  = PKB + PK_EGC + (ei * 6 + 0) * 4096;
    const bf16* pk  = PKB + PK_EGC + (ei * 6 + 1) * 4096;
    const bf16* pv  = PKB + PK_EGC + (ei * 6 + 2) * 4096;
    const bf16* pe  = PKB + PK_EGC + (ei * 6 + 3) * 4096;
    const bf16* pev = PKB + PK_EGC + (ei * 6 + 4) * 4096;
    const bf16* po  = PKB + PK_EGC + (ei * 6 + 5) * 4096;
    gemm(nf, Nn, 64, pq, F(pb + 13), nullptr, nullptr, nullptr, Q,  MODE_BIAS);
    gemm(nf, Nn, 64, pk, F(pb + 5),  nullptr, nullptr, nullptr, Kf, MODE_BIAS);
    gemm(nf, Nn, 64, pv, F(pb + 15), nullptr, nullptr, nullptr, V,  MODE_BIAS);
    gemm(ef, Ne, 64, pe, F(pb + 1),  nullptr, nullptr, nullptr, Mb, MODE_BIAS);
    zero(Agg, (size_t)Nn * 64);
    egc_attn_kernel<<<(Ne + 3) / 4, 256, 0, stream>>>(Q, Kf, V, Mb, gs, gd, Ne, Agg);
    // y_out = lin_ev(efeat) + silu(ln_e(m))    (in place over efeat)
    gemm(ef, Ne, 64, pev, F(pb + 3), F(pb + 7), F(pb + 6), Mb, ef, MODE_ADD_LNSILU);
    // x_out = lin_out(nfeat) + silu(ln_n(agg)) (in place over nfeat)
    gemm(nf, Nn, 64, po, F(pb + 11), F(pb + 9), F(pb + 8), Agg, nf, MODE_ADD_LNSILU);
  };

  // ALIGNN layer 0
  run_egc(18, 0, X,  B1, src,  dst,  N, E, AQ, AK, AV, B2, AGG);
  run_egc(2,  1, B1, C1, lsrc, ldst, E, L, B2, B4, B5, C2, B6);
  // ALIGNN layer 1
  run_egc(50, 2, X,  B1, src,  dst,  N, E, AQ, AK, AV, B2, AGG);
  run_egc(34, 3, B1, C1, lsrc, ldst, E, L, B2, B4, B5, C2, B6);
  // GCN layers
  run_egc(88,  4, X, B1, src, dst, N, E, AQ, AK, AV, B2, AGG);
  run_egc(104, 5, X, B1, src, dst, N, E, AQ, AK, AV, B2, AGG);

  // ---- Readout ----------------------------------------------------------
  pool_kernel<<<512, 256, 0, stream>>>(X, H, N);
  final_kernel<<<1, 64, 0, stream>>>(H, F(86), F(87), (float*)d_out, 1.0f / (float)N);
}
